// SoftmaxCausalAttention_46540265619721
// MI455X (gfx1250) — matrix operand order repaired and checked by reading, not compiled, not run on hardware
//
#include <hip/hip_runtime.h>

// ---------------------------------------------------------------------------
// Causal multi-head attention for MI455X (gfx1250, wave32, WMMA bf16 path).
//   q,k,v = x@W{q,k,v}^T + b   -> bf16 workspace (V stored head-transposed)
//   flash-attention (online softmax, causal) -> bf16 workspace
//   out = attn@Wo^T + bo       -> f32 d_out
// GEMMs: 128x128 block / 8 waves of 32x64, K-step 32, double-buffered LDS,
// B-fragments preloaded so each K-step is one ds-wait + 8 back-to-back WMMAs.
// ---------------------------------------------------------------------------

typedef __bf16 bf16;
typedef __attribute__((ext_vector_type(4)))  __bf16 v4bf;
typedef __attribute__((ext_vector_type(8)))  __bf16 v8bf;
typedef __attribute__((ext_vector_type(16))) __bf16 v16bf;
typedef __attribute__((ext_vector_type(4)))  float  v4f;
typedef __attribute__((ext_vector_type(8)))  float  v8f;

#define BATCH   2
#define SLEN    2048
#define DMODEL  1024
#define NHEAD   16
#define DHEAD   64

// ---- helpers ---------------------------------------------------------------

__device__ __forceinline__ bf16 f2bf(float f) {
  union { float f; unsigned u; } v; v.f = f;
  unsigned r = v.u + 0x7fffu + ((v.u >> 16) & 1u);   // round-to-nearest-even
  unsigned short h = (unsigned short)(r >> 16);
  return __builtin_bit_cast(bf16, h);
}

__device__ __forceinline__ v16bf cat16(v8bf a, v8bf b) {
  v16bf r;
#pragma unroll
  for (int j = 0; j < 8; ++j) { r[j] = a[j]; r[j + 8] = b[j]; }
  return r;
}

// A fragment, 16x32 bf16.  lane&15 = row m; lanes 0-15 hold K=0..15,
// lanes 16-31 hold K=16..31 (same K order as load_b_frag).
__device__ __forceinline__ v16bf load_a_frag(const bf16* p, int ld, int lane) {
  const int m = lane & 15, hi = lane >> 4;
  const bf16* row = p + m * ld + hi * 16;
  return cat16(*(const v8bf*)row, *(const v8bf*)(row + 8));
}

// B fragment, 32x16 bf16.  lane&15 = col n; lanes 0-15 K=0..15, 16-31 K=16..31.
// Element (k,n) is read from p[n*ld + k]  (p indexed as [n][k]).
__device__ __forceinline__ v16bf load_b_frag(const bf16* p, int ld, int lane) {
  const int n = lane & 15, hi = lane >> 4;
  const bf16* row = p + n * ld + hi * 16;
  return cat16(*(const v8bf*)row, *(const v8bf*)(row + 8));
}

__device__ __forceinline__ v8f wmma_bf(v16bf a, v16bf b, v8f c) {
  // v_wmma_f32_16x16x32_bf16  (8-arg probe-confirmed signature)
  return __builtin_amdgcn_wmma_f32_16x16x32_bf16(false, a, false, b,
                                                 (short)0, c, false, false);
}

// ---- fp32 -> bf16 conversion ----------------------------------------------

__global__ __launch_bounds__(256)
void cvt_f32_to_bf16(const float* __restrict__ in, bf16* __restrict__ out,
                     int n4) {
  const int i = blockIdx.x * 256 + threadIdx.x;
  if (i < n4) {
    v4f f = *(const v4f*)(in + (long)i * 4);
    v4bf o;
#pragma unroll
    for (int j = 0; j < 4; ++j) o[j] = f2bf(f[j]);
    *(v4bf*)(out + (long)i * 4) = o;
  }
}

// ---- GEMM: Y = A @ W^T + bias ---------------------------------------------
// A: [M,K] bf16 row-major.  W: [N,K] bf16 row-major (so W^T columns are W rows,
// which is exactly the contiguous-along-K layout the B-fragment wants).
// mode 0: Y bf16 [M,N];  mode 1: Y bf16 head-transposed [B, N, S];
// mode 2: Y f32 [M,N].

__global__ __launch_bounds__(256)
void gemm_xwT(const bf16* __restrict__ A, const bf16* __restrict__ W,
              const float* __restrict__ bias, void* __restrict__ Y,
              int M, int N, int Kd, int mode) {
  __shared__ bf16 sA[2][128 * 32];
  __shared__ bf16 sW[2][128 * 32];

  const int tid  = threadIdx.x;
  const int lane = tid & 31, wv = tid >> 5;
  const int wm = wv & 3, wn = wv >> 2;               // 4 x 2 wave grid
  const long bm = (long)blockIdx.y * 128, bn = (long)blockIdx.x * 128;
  const int lr = tid >> 1, lc = (tid & 1) * 16;

  v8f acc[2][4];
#pragma unroll
  for (int i = 0; i < 2; ++i)
#pragma unroll
    for (int t = 0; t < 4; ++t)
#pragma unroll
      for (int r = 0; r < 8; ++r) acc[i][t][r] = 0.f;

  const bf16* ga = A + (bm + lr) * Kd + lc;
  const bf16* gw = W + (bn + lr) * Kd + lc;

  // prologue: stage K-slab 0 into buffer 0
  *(v8bf*)&sA[0][lr * 32 + lc]     = *(const v8bf*)(ga);
  *(v8bf*)&sA[0][lr * 32 + lc + 8] = *(const v8bf*)(ga + 8);
  *(v8bf*)&sW[0][lr * 32 + lc]     = *(const v8bf*)(gw);
  *(v8bf*)&sW[0][lr * 32 + lc + 8] = *(const v8bf*)(gw + 8);

  int buf = 0;
  for (int k0 = 0; k0 < Kd; k0 += 32) {
    __syncthreads();                 // buf stores visible; prev reads done
    const int nxt = buf ^ 1;
    if (k0 + 32 < Kd) {              // stage next slab under this slab's WMMAs
      const bf16* ga2 = ga + k0 + 32;
      const bf16* gw2 = gw + k0 + 32;
      *(v8bf*)&sA[nxt][lr * 32 + lc]     = *(const v8bf*)(ga2);
      *(v8bf*)&sA[nxt][lr * 32 + lc + 8] = *(const v8bf*)(ga2 + 8);
      *(v8bf*)&sW[nxt][lr * 32 + lc]     = *(const v8bf*)(gw2);
      *(v8bf*)&sW[nxt][lr * 32 + lc + 8] = *(const v8bf*)(gw2 + 8);
      if (k0 + 64 < Kd) {            // global_prefetch_b8, near-temporal
        __builtin_prefetch(ga + k0 + 64, 0, 3);
        __builtin_prefetch(gw + k0 + 64, 0, 3);
      }
    }
    // preload both A-frags and all four B-frags, then 8 back-to-back WMMAs
    v16bf a0 = load_a_frag(&sA[buf][(wm * 32 +  0) * 32], 32, lane);
    v16bf a1 = load_a_frag(&sA[buf][(wm * 32 + 16) * 32], 32, lane);
    v16bf bt[4];
#pragma unroll
    for (int t = 0; t < 4; ++t)
      bt[t] = load_b_frag(&sW[buf][(wn * 64 + t * 16) * 32], 32, lane);
#pragma unroll
    for (int t = 0; t < 4; ++t) acc[0][t] = wmma_bf(a0, bt[t], acc[0][t]);
#pragma unroll
    for (int t = 0; t < 4; ++t) acc[1][t] = wmma_bf(a1, bt[t], acc[1][t]);
    buf = nxt;
  }

  const int nl = lane & 15, hi = lane >> 4;
#pragma unroll
  for (int i = 0; i < 2; ++i)
#pragma unroll
    for (int t = 0; t < 4; ++t) {
      const long ng = bn + wn * 64 + t * 16 + nl;
      const float bv = bias[ng];
#pragma unroll
      for (int r = 0; r < 8; ++r) {
        const long mg  = bm + wm * 32 + i * 16 + r + 8 * hi;
        const float v = acc[i][t][r] + bv;
        if (mode == 0) {
          ((bf16*)Y)[mg * N + ng] = f2bf(v);
        } else if (mode == 1) {
          const long bb = mg >> 11, ss = mg & (SLEN - 1);   // S = 2048
          ((bf16*)Y)[(bb * N + ng) * SLEN + ss] = f2bf(v);
        } else {
          ((float*)Y)[mg * N + ng] = v;
        }
      }
    }
}

// ---- Flash attention (causal, online softmax) ------------------------------
// Q,K: bf16 [B*S, D] (head h occupies cols h*64..h*64+63).
// Vt:  bf16 [B, D, S] (head-transposed so PV B-frags are contiguous).
// O:   bf16 [B*S, D].
// grid = (S/128, B*H); 8 waves; wave w owns q rows q0 = blk*128 + w*16.

__global__ __launch_bounds__(256)
void flash_attn(const bf16* __restrict__ Q, const bf16* __restrict__ K,
                const bf16* __restrict__ Vt, bf16* __restrict__ O) {
  __shared__ bf16 sP[8 * 16 * 64];           // per-wave P tile (16 x 64)

  const int lane = threadIdx.x & 31;
  const int wv   = threadIdx.x >> 5;
  const int b = blockIdx.y >> 4, h = blockIdx.y & 15;
  const int q0 = blockIdx.x * 128 + wv * 16;
  const int nl = lane & 15, hi = lane >> 4;

  const bf16* qp = Q + ((long)(b * SLEN + q0)) * DMODEL + h * DHEAD;
  const v16bf aq0 = load_a_frag(qp,      DMODEL, lane);   // dh 0..31
  const v16bf aq1 = load_a_frag(qp + 32, DMODEL, lane);   // dh 32..63

  v8f o[4];
  float mrun[8], lrun[8];
#pragma unroll
  for (int r = 0; r < 8; ++r) { mrun[r] = -1e30f; lrun[r] = 0.f; }
#pragma unroll
  for (int d = 0; d < 4; ++d)
#pragma unroll
    for (int r = 0; r < 8; ++r) o[d][r] = 0.f;

  bf16* pp = sP + wv * (16 * 64);
  const int kend = ((q0 + 16 + 63) >> 6) << 6;   // per-wave causal bound

  for (int kb = 0; kb < kend; kb += 64) {
    const bool more = (kb + 64 < kend);
    // ---- scores S = Q K^T / sqrt(Dh), 16 q-rows x 64 k-cols --------------
    v8f s[4];
#pragma unroll
    for (int t = 0; t < 4; ++t) {
      const bf16* kp = K + ((long)(b * SLEN + kb + t * 16)) * DMODEL + h * DHEAD;
      const v16bf bk0 = load_b_frag(kp,      DMODEL, lane);
      const v16bf bk1 = load_b_frag(kp + 32, DMODEL, lane);
      if (more) __builtin_prefetch(kp + (long)64 * DMODEL, 0, 3);
      v8f a;
#pragma unroll
      for (int r = 0; r < 8; ++r) a[r] = 0.f;
      a = wmma_bf(aq0, bk0, a);
      a = wmma_bf(aq1, bk1, a);
      s[t] = a;
    }
    // scale + causal mask (D-frag: m = r + 8*hi, n = lane&15)
#pragma unroll
    for (int t = 0; t < 4; ++t) {
      const int kg = kb + t * 16 + nl;
#pragma unroll
      for (int r = 0; r < 8; ++r) {
        const int qg = q0 + r + 8 * hi;
        const float v = s[t][r] * 0.125f;            // 1/sqrt(64)
        s[t][r] = (kg <= qg) ? v : -__builtin_inff();
      }
    }
    // ---- online softmax --------------------------------------------------
    float rmax[8], rsum[8], fac[8];
#pragma unroll
    for (int r = 0; r < 8; ++r)
      rmax[r] = fmaxf(fmaxf(s[0][r], s[1][r]), fmaxf(s[2][r], s[3][r]));
#pragma unroll
    for (int mk = 1; mk <= 8; mk <<= 1)      // reduce within 16-lane half
#pragma unroll
      for (int r = 0; r < 8; ++r)
        rmax[r] = fmaxf(rmax[r], __shfl_xor(rmax[r], mk, 32));
#pragma unroll
    for (int r = 0; r < 8; ++r) {
      const float mn = fmaxf(mrun[r], rmax[r]);
      fac[r] = __expf(mrun[r] - mn);
      mrun[r] = mn;
      rsum[r] = 0.f;
    }
#pragma unroll
    for (int t = 0; t < 4; ++t)
#pragma unroll
      for (int r = 0; r < 8; ++r) {
        const float e = __expf(s[t][r] - mrun[r]);
        s[t][r] = e;
        rsum[r] += e;
      }
#pragma unroll
    for (int mk = 1; mk <= 8; mk <<= 1)
#pragma unroll
      for (int r = 0; r < 8; ++r)
        rsum[r] += __shfl_xor(rsum[r], mk, 32);
#pragma unroll
    for (int r = 0; r < 8; ++r) lrun[r] = lrun[r] * fac[r] + rsum[r];
#pragma unroll
    for (int d = 0; d < 4; ++d)
#pragma unroll
      for (int r = 0; r < 8; ++r) o[d][r] *= fac[r];

    // ---- re-layout P: D-frag -> A-frag via per-wave LDS tile -------------
#pragma unroll
    for (int t = 0; t < 4; ++t)
#pragma unroll
      for (int r = 0; r < 8; ++r)
        pp[(r + 8 * hi) * 64 + t * 16 + nl] = f2bf(s[t][r]);
    asm volatile("s_wait_dscnt 0" ::: "memory");   // wave-internal LDS order
    const v16bf ap0 = load_a_frag(pp,      64, lane);   // k-cols 0..31
    const v16bf ap1 = load_a_frag(pp + 32, 64, lane);   // k-cols 32..63

    // ---- O += P V  (B-frags contiguous thanks to Vt layout) --------------
#pragma unroll
    for (int d = 0; d < 4; ++d) {
      const bf16* vp = Vt + ((long)(b * DMODEL + h * DHEAD + d * 16)) * SLEN + kb;
      const v16bf bv0 = load_b_frag(vp,      SLEN, lane);
      const v16bf bv1 = load_b_frag(vp + 32, SLEN, lane);
      if (more) __builtin_prefetch(vp + 64, 0, 3);
      o[d] = wmma_bf(ap0, bv0, o[d]);
      o[d] = wmma_bf(ap1, bv1, o[d]);
    }
  }

  // ---- normalize + store ---------------------------------------------------
#pragma unroll
  for (int r = 0; r < 8; ++r) lrun[r] = 1.f / lrun[r];
#pragma unroll
  for (int d = 0; d < 4; ++d)
#pragma unroll
    for (int r = 0; r < 8; ++r) {
      const int sg = q0 + r + 8 * hi;
      O[((long)(b * SLEN + sg)) * DMODEL + h * DHEAD + d * 16 + nl] =
          f2bf(o[d][r] * lrun[r]);
    }
}

// ---- host-side orchestration ----------------------------------------------

extern "C" void kernel_launch(void* const* d_in, const int* in_sizes, int n_in,
                              void* d_out, int out_size, void* d_ws,
                              size_t ws_size, hipStream_t stream) {
  (void)in_sizes; (void)n_in; (void)out_size; (void)ws_size;

  const float* x  = (const float*)d_in[0];
  const float* Wq = (const float*)d_in[1];
  const float* bq = (const float*)d_in[2];
  const float* Wk = (const float*)d_in[3];
  const float* bk = (const float*)d_in[4];
  const float* Wv = (const float*)d_in[5];
  const float* bv = (const float*)d_in[6];
  const float* Wo = (const float*)d_in[7];
  const float* bo = (const float*)d_in[8];

  const long MTOK = (long)BATCH * SLEN;           // 4096 rows
  const long XB   = MTOK * DMODEL;                // 4 Mi elems
  const long WB   = (long)DMODEL * DMODEL;        // 1 Mi elems

  char* w = (char*)d_ws;
  bf16* xb  = (bf16*)w; w += XB * 2;
  bf16* Wqb = (bf16*)w; w += WB * 2;
  bf16* Wkb = (bf16*)w; w += WB * 2;
  bf16* Wvb = (bf16*)w; w += WB * 2;
  bf16* Wob = (bf16*)w; w += WB * 2;
  bf16* Qb  = (bf16*)w; w += XB * 2;
  bf16* Kb  = (bf16*)w; w += XB * 2;
  bf16* Vt  = (bf16*)w; w += XB * 2;              // [B, D, S]
  bf16* Ab  = (bf16*)w; w += XB * 2;

  dim3 blk(256);
  cvt_f32_to_bf16<<<(unsigned)((XB / 4 + 255) / 256), blk, 0, stream>>>(x,  xb,  (int)(XB / 4));
  cvt_f32_to_bf16<<<(unsigned)((WB / 4 + 255) / 256), blk, 0, stream>>>(Wq, Wqb, (int)(WB / 4));
  cvt_f32_to_bf16<<<(unsigned)((WB / 4 + 255) / 256), blk, 0, stream>>>(Wk, Wkb, (int)(WB / 4));
  cvt_f32_to_bf16<<<(unsigned)((WB / 4 + 255) / 256), blk, 0, stream>>>(Wv, Wvb, (int)(WB / 4));
  cvt_f32_to_bf16<<<(unsigned)((WB / 4 + 255) / 256), blk, 0, stream>>>(Wo, Wob, (int)(WB / 4));

  dim3 gg(DMODEL / 128, (unsigned)(MTOK / 128));  // (8, 32)
  gemm_xwT<<<gg, blk, 0, stream>>>(xb, Wqb, bq, Qb, (int)MTOK, DMODEL, DMODEL, 0);
  gemm_xwT<<<gg, blk, 0, stream>>>(xb, Wkb, bk, Kb, (int)MTOK, DMODEL, DMODEL, 0);
  gemm_xwT<<<gg, blk, 0, stream>>>(xb, Wvb, bv, Vt, (int)MTOK, DMODEL, DMODEL, 1);

  dim3 ga(SLEN / 128, BATCH * NHEAD);             // (16, 32)
  flash_attn<<<ga, blk, 0, stream>>>(Qb, Kb, Vt, Ab);

  gemm_xwT<<<gg, blk, 0, stream>>>(Ab, Wob, bo, d_out, (int)MTOK, DMODEL, DMODEL, 2);
}
